// AUGRU_69569880261404
// MI455X (gfx1250) — matrix operand order repaired and checked by reading, not compiled, not run on hardware
//
#include <hip/hip_runtime.h>
#include <hip/hip_bf16.h>
#include <math.h>

// ---------------------------------------------------------------------------
// AUGRU fused cell for MI455X (gfx1250, wave32, WMMA).
// Six 8192x512x512 GEMMs with v_wmma_f32_16x16x32_bf16 using the bf16x3
// split (hi*hi + lo*hi + hi*lo) for ~fp32 accuracy, fused sigmoid/tanh/
// attention epilogue. Compute-bound (AI ~478 FLOP/B); weights + bf16
// activations (~40MB) are L2-resident (192MB).
//
// Round-3 changes:
//  * All load addresses are hoisted, loop-invariant 32-bit per-lane element
//    offsets off uniform (SGPR) base pointers -> backend can use the
//    saddr+voffset global-load form; K advance folds into the 24-bit
//    instruction offset inside the unrolled body. This removes the
//    per-iteration v_add_nc_u64 ops that were overwriting WMMA source
//    registers (WMMA->VALU WAR hazard => v_nop padding) and drops ~32 live
//    VGPRs (back under the 256 direct-address limit).
//  * #pragma unroll 2 on the K loop.
// ---------------------------------------------------------------------------

typedef __attribute__((ext_vector_type(8)))  float  v8f;
typedef __attribute__((ext_vector_type(8)))  __bf16 v8bf;
typedef __attribute__((ext_vector_type(16))) __bf16 v16bf;

#define BDIM   8192
#define KDIM   512
#define NDIM   512

// ---------------- prep: fp32 -> bf16 hi/lo split (row major) ----------------
__global__ void split_bf16_kernel(const float* __restrict__ in,
                                  __bf16* __restrict__ hi,
                                  __bf16* __restrict__ lo, int n) {
  int i = blockIdx.x * blockDim.x + threadIdx.x;
  if (i < n) {
    float v = in[i];
    __bf16 h = (__bf16)v;
    hi[i] = h;
    lo[i] = (__bf16)(v - (float)h);
  }
}

// ------ prep: weight fp32 [K][N] row-major -> bf16 hi/lo transposed [N][K] --
__global__ void transpose_split_bf16_kernel(const float* __restrict__ W,
                                            __bf16* __restrict__ hi,
                                            __bf16* __restrict__ lo) {
  int i = blockIdx.x * blockDim.x + threadIdx.x;   // KDIM*NDIM threads
  int k = i >> 9;          // row (K)
  int n = i & 511;         // col (N)
  float v = W[i];          // coalesced read
  __bf16 h = (__bf16)v;
  hi[(size_t)n * KDIM + k] = h;
  lo[(size_t)n * KDIM + k] = (__bf16)(v - (float)h);
}

// ---------------- fragment loaders (contiguous k order, same in A and B) --------------
// A (16x32 bf16): lane L -> row = L&15 ; K runs {kb..kb+15}
// with kb = 16*(L>>4): two contiguous 8-element (16B) loads.
__device__ __forceinline__ v16bf load_a_frag(const __bf16* __restrict__ base,
                                             unsigned off) {
  v8bf a0 = *(const v8bf*)(base + off);
  v8bf a1 = *(const v8bf*)(base + off + 8);
  return __builtin_shufflevector(a0, a1, 0, 1, 2, 3, 4, 5, 6, 7,
                                         8, 9, 10, 11, 12, 13, 14, 15);
}

// B (32x16 bf16): lane L -> col = L&15, K base = 16*(L>>4); with weights
// stored N-major ([N][K]) this is 16 contiguous bf16 (one 32B load).
__device__ __forceinline__ v16bf load_b_frag(const __bf16* __restrict__ base,
                                             unsigned off) {
  return *(const v16bf*)(base + off);
}

__device__ __forceinline__ v8f wmma1(v16bf a, v16bf b, v8f c) {
  return __builtin_amdgcn_wmma_f32_16x16x32_bf16(false, a, false, b,
                                                 (short)0, c, false, false);
}

__device__ __forceinline__ float sigmoidf_(float x) {
  return 1.0f / (1.0f + __expf(-x));
}

// One bf16x3 "pass": 6 WMMAs cycling through 3 gate accumulators x 2 M-tiles.
// Same accumulator is revisited only every 6 WMMAs -> no hazard NOPs needed.
#define PASS6(AF0, AF1, B0, B1, B2, ACC0, ACC1, ACC2)            \
  do {                                                           \
    ACC0[0][ni] = wmma1(AF0, B0, ACC0[0][ni]);                   \
    ACC0[1][ni] = wmma1(AF1, B0, ACC0[1][ni]);                   \
    ACC1[0][ni] = wmma1(AF0, B1, ACC1[0][ni]);                   \
    ACC1[1][ni] = wmma1(AF1, B1, ACC1[1][ni]);                   \
    ACC2[0][ni] = wmma1(AF0, B2, ACC2[0][ni]);                   \
    ACC2[1][ni] = wmma1(AF1, B2, ACC2[1][ni]);                   \
  } while (0)

// ----------------------------- fused main kernel ----------------------------
// Block = 256 threads = 8 waves (4 M-waves x 2 N-waves), tile = 128(M) x 64(N)
// Each wave: 32x32 outputs = 2x2 WMMA tiles x 4 accumulator sets.
__global__ __launch_bounds__(256)
void augru_fused_kernel(const float* __restrict__ h_in,
                        const float* __restrict__ att,
                        const float* __restrict__ biasR,
                        const float* __restrict__ biasU,
                        const float* __restrict__ biasH,
                        const __bf16* __restrict__ xh, const __bf16* __restrict__ xl,
                        const __bf16* __restrict__ hh, const __bf16* __restrict__ hl,
                        const __bf16* __restrict__ wxr_h, const __bf16* __restrict__ wxr_l,
                        const __bf16* __restrict__ whr_h, const __bf16* __restrict__ whr_l,
                        const __bf16* __restrict__ wxu_h, const __bf16* __restrict__ wxu_l,
                        const __bf16* __restrict__ whu_h, const __bf16* __restrict__ whu_l,
                        const __bf16* __restrict__ wxh_h, const __bf16* __restrict__ wxh_l,
                        const __bf16* __restrict__ whh_h, const __bf16* __restrict__ whh_l,
                        float* __restrict__ out) {
  const int lane  = threadIdx.x & 31;
  const int wave  = threadIdx.x >> 5;
  const int waveM = wave >> 1;          // 0..3
  const int waveN = wave & 1;           // 0..1
  const int mW = blockIdx.y * 128 + waveM * 32;
  const int nW = blockIdx.x * 64  + waveN * 32;
  const int lm   = lane & 15;
  const int ksel = lane >> 4;           // 0 or 1

  v8f accR[2][2] = {};
  v8f accU[2][2] = {};
  v8f accC[2][2] = {};  // x @ Wx_h
  v8f accD[2][2] = {};  // h @ Wh_h

  // Loop-invariant 32-bit per-lane element offsets (K advance comes from the
  // uniform loop counter and folds into instruction immediates on unroll).
  unsigned aoff[2], boff[2];
#pragma unroll
  for (int mi = 0; mi < 2; ++mi)
    aoff[mi] = (unsigned)(mW + mi * 16 + lm) * KDIM + 16u * ksel;
#pragma unroll
  for (int ni = 0; ni < 2; ++ni)
    boff[ni] = (unsigned)(nW + ni * 16 + lm) * KDIM + 16u * ksel;

#pragma unroll 2
  for (unsigned kb = 0; kb < KDIM; kb += 32) {
    v16bf afh[2], afl[2];

    // ---- x side: Wx_r -> accR, Wx_u -> accU, Wx_h -> accC ----
#pragma unroll
    for (int mi = 0; mi < 2; ++mi) {
      afh[mi] = load_a_frag(xh, aoff[mi] + kb);
      afl[mi] = load_a_frag(xl, aoff[mi] + kb);
    }
#pragma unroll
    for (int ni = 0; ni < 2; ++ni) {
      const unsigned bo = boff[ni] + kb;
      v16bf b0 = load_b_frag(wxr_h, bo);
      v16bf b1 = load_b_frag(wxu_h, bo);
      v16bf b2 = load_b_frag(wxh_h, bo);
      PASS6(afh[0], afh[1], b0, b1, b2, accR, accU, accC);   // hi * hi
      PASS6(afl[0], afl[1], b0, b1, b2, accR, accU, accC);   // lo * hi
      b0 = load_b_frag(wxr_l, bo);
      b1 = load_b_frag(wxu_l, bo);
      b2 = load_b_frag(wxh_l, bo);
      PASS6(afh[0], afh[1], b0, b1, b2, accR, accU, accC);   // hi * lo
    }

    // ---- h side: Wh_r -> accR, Wh_u -> accU, Wh_h -> accD ----
#pragma unroll
    for (int mi = 0; mi < 2; ++mi) {
      afh[mi] = load_a_frag(hh, aoff[mi] + kb);
      afl[mi] = load_a_frag(hl, aoff[mi] + kb);
    }
#pragma unroll
    for (int ni = 0; ni < 2; ++ni) {
      const unsigned bo = boff[ni] + kb;
      v16bf b0 = load_b_frag(whr_h, bo);
      v16bf b1 = load_b_frag(whu_h, bo);
      v16bf b2 = load_b_frag(whh_h, bo);
      PASS6(afh[0], afh[1], b0, b1, b2, accR, accU, accD);   // hi * hi
      PASS6(afl[0], afl[1], b0, b1, b2, accR, accU, accD);   // lo * hi
      b0 = load_b_frag(whr_l, bo);
      b1 = load_b_frag(whu_l, bo);
      b2 = load_b_frag(whh_l, bo);
      PASS6(afh[0], afh[1], b0, b1, b2, accR, accU, accD);   // hi * lo
    }
  }

  // ---- fused epilogue: C/D tile layout: row = j + 8*ksel, col = lm ----
#pragma unroll
  for (int mi = 0; mi < 2; ++mi) {
#pragma unroll
    for (int ni = 0; ni < 2; ++ni) {
      const int n = nW + ni * 16 + lm;
      const float brv = biasR[n];
      const float buv = biasU[n];
      const float bhv = biasH[n];
#pragma unroll
      for (int j = 0; j < 8; ++j) {
        const int m = mW + mi * 16 + 8 * ksel + j;
        const float a  = att[m];
        const float hv = h_in[(size_t)m * NDIM + n];
        const float rv = sigmoidf_(accR[mi][ni][j] + brv);
        const float uv = sigmoidf_(accU[mi][ni][j] + buv);
        const float cv = tanhf(accC[mi][ni][j] + bhv + rv * accD[mi][ni][j]);
        const float ua = a * uv;
        out[(size_t)m * NDIM + n] = (1.0f - ua) * hv + ua * cv;
      }
    }
  }
}

// ---------------------------------------------------------------------------
extern "C" void kernel_launch(void* const* d_in, const int* in_sizes, int n_in,
                              void* d_out, int out_size, void* d_ws, size_t ws_size,
                              hipStream_t stream) {
  (void)in_sizes; (void)n_in; (void)out_size; (void)ws_size;

  const float* input_x = (const float*)d_in[0];
  const float* input_h = (const float*)d_in[1];
  const float* att     = (const float*)d_in[2];
  const float* Wx_r    = (const float*)d_in[3];
  const float* b_r     = (const float*)d_in[4];
  const float* Wh_r    = (const float*)d_in[5];
  const float* Wx_u    = (const float*)d_in[6];
  const float* b_u     = (const float*)d_in[7];
  const float* Wh_u    = (const float*)d_in[8];
  const float* Wx_h    = (const float*)d_in[9];
  const float* b_h     = (const float*)d_in[10];
  const float* Wh_h    = (const float*)d_in[11];
  float* out = (float*)d_out;

  // Workspace layout (bf16 elements): x_hi, x_lo, h_hi, h_lo, then 12 weight
  // arrays (hi/lo per weight, transposed N-major). ~40 MB total.
  __bf16* w = (__bf16*)d_ws;
  const size_t NA = (size_t)BDIM * KDIM;   // 8192*512
  const size_t NW = (size_t)KDIM * NDIM;   // 512*512
  __bf16* xh = w + 0 * NA;
  __bf16* xl = w + 1 * NA;
  __bf16* hh = w + 2 * NA;
  __bf16* hl = w + 3 * NA;
  __bf16* wp = w + 4 * NA;
  __bf16* wptr[12];
  for (int i = 0; i < 12; ++i) wptr[i] = wp + (size_t)i * NW;
  // order: wxr_h, wxr_l, whr_h, whr_l, wxu_h, wxu_l, whu_h, whu_l,
  //        wxh_h, wxh_l, whh_h, whh_l

  // --- prep: activation splits ---
  {
    int threads = 256;
    int blocks = (int)((NA + threads - 1) / threads);
    split_bf16_kernel<<<blocks, threads, 0, stream>>>(input_x, xh, xl, (int)NA);
    split_bf16_kernel<<<blocks, threads, 0, stream>>>(input_h, hh, hl, (int)NA);
  }
  // --- prep: weight transpose + split ---
  {
    const float* Wsrc[6] = {Wx_r, Wh_r, Wx_u, Wh_u, Wx_h, Wh_h};
    int threads = 256;
    int blocks = (int)(NW / threads);
    for (int i = 0; i < 6; ++i) {
      transpose_split_bf16_kernel<<<blocks, threads, 0, stream>>>(
          Wsrc[i], wptr[2 * i], wptr[2 * i + 1]);
    }
  }

  // --- fused AUGRU: grid = (N tiles, M tiles) = (512/64, 8192/128) ---
  dim3 grid(NDIM / 64, BDIM / 128);
  dim3 block(256);
  augru_fused_kernel<<<grid, block, 0, stream>>>(
      input_h, att, b_r, b_u, b_h,
      xh, xl, hh, hl,
      wptr[0], wptr[1], wptr[2], wptr[3],
      wptr[4], wptr[5], wptr[6], wptr[7],
      wptr[8], wptr[9], wptr[10], wptr[11],
      out);
}
